// MSVA_SelfAttention_52948356825145
// MI455X (gfx1250) — compile-verified, matrix operand order checked by reading, not run on hardware
//
#include <hip/hip_runtime.h>
#include <cstdint>
#include <cstddef>

// MI455X / gfx1250: wave32, WMMA bf16 16x16x32, f32 accumulate.

#define DEVINL __device__ __forceinline__

typedef __attribute__((ext_vector_type(16))) __bf16        bf16x16;
typedef __attribute__((ext_vector_type(8)))  float         floatx8;
typedef __attribute__((ext_vector_type(4)))  unsigned int  uint32x4;

constexpr int   B_ = 4;
constexpr int   N_ = 4096;
constexpr int   D_ = 1024;
constexpr int   AP = 256;          // apperture (band half-width)
constexpr float QS = 0.06f;
constexpr int   MAXT = 34;         // max 16-col tiles spanning one 16-row band

// ---------- helpers ----------

DEVINL unsigned short f2bf(float f) {
  union { float f; unsigned u; } x; x.f = f;
  unsigned r = x.u + 0x7FFFu + ((x.u >> 16) & 1u);   // round-to-nearest-even
  return (unsigned short)(r >> 16);
}

union FragU {
  bf16x16        v;
  uint32x4       q[2];
  unsigned short h[16];
};

// Load a 16x32 bf16 fragment from row-major memory (rows on lanes).
// Matches ISA 7.12.2 16-bit A layout; B (K x N) uses the identical pattern
// with lane = N column, so the same loader serves both operands of C = A*B^T.
DEVINL bf16x16 load_frag_rm(const unsigned short* base, int ld, int k0, int lane) {
  int row = lane & 15;
  int kg  = (lane >> 4) & 1;
  const unsigned short* p = base + (size_t)row * ld + (size_t)(k0 + kg * 8);
  FragU u;
  u.q[0] = *(const uint32x4*)(p);        // K = k0+kg*8 + 0..7
  u.q[1] = *(const uint32x4*)(p + 16);   // K = k0+kg*8 + 16..23
  return u.v;
}

DEVINL floatx8 wmma_bf16(bf16x16 a, bf16x16 b, floatx8 c) {
  return __builtin_amdgcn_wmma_f32_16x16x32_bf16(
      /*neg_a=*/false, a, /*neg_b=*/false, b,
      /*c_mod=*/(short)0, c, /*reuse_a=*/false, /*reuse_b=*/false);
}

// ---------- kernels ----------

__global__ void f32_to_bf16_kernel(const float* __restrict__ src,
                                   unsigned short* __restrict__ dst, size_t n) {
  size_t i = (size_t)blockIdx.x * blockDim.x + threadIdx.x;
  size_t stride = (size_t)gridDim.x * blockDim.x;
  for (; i < n; i += stride) dst[i] = f2bf(src[i]);
}

__global__ void zero_f32_kernel(float* __restrict__ p, size_t n) {
  size_t i = (size_t)blockIdx.x * blockDim.x + threadIdx.x;
  size_t stride = (size_t)gridDim.x * blockDim.x;
  for (; i < n; i += stride) p[i] = 0.0f;
}

// C = A(Mtot x K) * Bw(Nc x K)^T,  A/Bw bf16 row-major.
// mode 0: bf16 row-major C, scaled        (Q, K projections)
// mode 1: bf16 transposed-per-batch C -> Ct[b, n_col, row%N]   (V -> Vt)
// mode 2: f32 row-major C                 (final output projection)
// block: 256 threads = 8 waves in 2(M) x 4(N); block tile 64x128; wave tile 32x32.
__global__ void gemm_wmma_kernel(const unsigned short* __restrict__ A,
                                 const unsigned short* __restrict__ Bw,
                                 void* __restrict__ Cout,
                                 int Mtot, int Nc, int K,
                                 float scale, int mode) {
  int lane = threadIdx.x & 31;
  int wid  = threadIdx.x >> 5;
  int wm = wid & 1, wn = wid >> 1;
  int m_base = blockIdx.x * 64 + wm * 32;
  int n_base = blockIdx.y * 128 + wn * 32;

  floatx8 acc[2][2] = {};
  const unsigned short* A0 = A + (size_t)m_base * K;
  const unsigned short* A1 = A + (size_t)(m_base + 16) * K;
  const unsigned short* B0 = Bw + (size_t)n_base * K;
  const unsigned short* B1 = Bw + (size_t)(n_base + 16) * K;

  for (int k0 = 0; k0 < K; k0 += 32) {
    bf16x16 a0 = load_frag_rm(A0, K, k0, lane);
    bf16x16 a1 = load_frag_rm(A1, K, k0, lane);
    bf16x16 b0 = load_frag_rm(B0, K, k0, lane);
    bf16x16 b1 = load_frag_rm(B1, K, k0, lane);
    acc[0][0] = wmma_bf16(a0, b0, acc[0][0]);
    acc[0][1] = wmma_bf16(a0, b1, acc[0][1]);
    acc[1][0] = wmma_bf16(a1, b0, acc[1][0]);
    acc[1][1] = wmma_bf16(a1, b1, acc[1][1]);
  }

  int col = lane & 15;
  int rhi = (lane >> 4) * 8;
  for (int ti = 0; ti < 2; ++ti)
    for (int tj = 0; tj < 2; ++tj)
      for (int i = 0; i < 8; ++i) {
        int gm = m_base + ti * 16 + rhi + i;
        int gn = n_base + tj * 16 + col;
        float v = acc[ti][tj][i] * scale;
        if (mode == 0) {
          ((unsigned short*)Cout)[(size_t)gm * Nc + gn] = f2bf(v);
        } else if (mode == 1) {
          int bb = gm >> 12;             // gm / N_
          int n  = gm & (N_ - 1);        // gm % N_
          ((unsigned short*)Cout)[((size_t)bb * Nc + gn) * N_ + n] = f2bf(v);
        } else {
          ((float*)Cout)[(size_t)gm * Nc + gn] = v;
        }
      }
}

// Banded logits + softmax. One wave (32 threads) per 16 query rows.
// logits[n,m] = sum_d Q[n,d]*K[m,d]; mask |n-m| >= AP; softmax over m;
// writes f32 att rows (out-of-band inside the tiled range gets exact 0).
__global__ void band_attn_kernel(const unsigned short* __restrict__ Qb,
                                 const unsigned short* __restrict__ Kb,
                                 float* __restrict__ att) {
  __shared__ float Ls[16][MAXT * 16];
  int b    = blockIdx.y;
  int r0   = blockIdx.x * 16;
  int lane = threadIdx.x;

  int ms = (r0 > (AP - 1)) ? ((r0 - (AP - 1)) & ~15) : 0;   // 16-aligned
  int me = min(N_, r0 + 15 + AP);                           // exclusive
  int nt = (me - ms + 15) >> 4;

  const unsigned short* Qbase = Qb + ((size_t)b * N_ + r0) * D_;

  for (int t = 0; t < nt; ++t) {
    int m0 = ms + t * 16;
    const unsigned short* Kbase = Kb + ((size_t)b * N_ + m0) * D_;
    floatx8 acc = {};
    for (int k0 = 0; k0 < D_; k0 += 32) {
      acc = wmma_bf16(load_frag_rm(Qbase, D_, k0, lane),
                      load_frag_rm(Kbase, D_, k0, lane), acc);
    }
    int col = lane & 15, rhi = (lane >> 4) * 8;
    for (int i = 0; i < 8; ++i) {
      int row = rhi + i;
      int d   = (r0 + row) - (m0 + col);
      float v = acc[i];
      if (d >= AP || d <= -AP) v = -__builtin_inff();
      Ls[row][t * 16 + col] = v;
    }
  }
  __syncthreads();

  int ncols = nt * 16;
  for (int r = 0; r < 16; ++r) {
    float mx = -__builtin_inff();
    for (int c = lane; c < ncols; c += 32) mx = fmaxf(mx, Ls[r][c]);
    for (int off = 16; off; off >>= 1) mx = fmaxf(mx, __shfl_xor(mx, off, 32));
    float s = 0.0f;
    for (int c = lane; c < ncols; c += 32) s += __expf(Ls[r][c] - mx);
    for (int off = 16; off; off >>= 1) s += __shfl_xor(s, off, 32);
    float inv = 1.0f / s;
    float* arow = att + ((size_t)b * N_ + (r0 + r)) * (size_t)N_ + ms;
    for (int c = lane; c < ncols; c += 32) arow[c] = __expf(Ls[r][c] - mx) * inv;
  }
}

// y[b,n,o] = sum_m att[b,m,n] * V[b,m,o]  ==  att^T * V, band-limited over m.
// A-fragment: strided f32 reads of att (transposed access) packed to bf16.
// B-fragment: contiguous reads from Vt[b,o,m]. 8 waves cover o in 128-col slabs.
__global__ void av_kernel(const float* __restrict__ att,
                          const unsigned short* __restrict__ Vt,
                          unsigned short* __restrict__ Y) {
  int b    = blockIdx.y;
  int r0   = blockIdx.x * 16;
  int wid  = threadIdx.x >> 5;
  int lane = threadIdx.x & 31;
  int o0   = wid * 128;

  int ms   = (r0 > (AP - 1)) ? ((r0 - (AP - 1)) & ~31) : 0;  // 32-aligned
  int me   = min(N_, r0 + 15 + AP);
  int me32 = (me + 31) & ~31;
  if (me32 > N_) me32 = N_;

  const float* attb = att + (size_t)b * N_ * (size_t)N_;
  floatx8 acc[8] = {};
  int row = lane & 15, kg = (lane >> 4) & 1;

  for (int m0 = ms; m0 < me32; m0 += 32) {
    FragU a;
#pragma unroll
    for (int idx = 0; idx < 16; ++idx) {
      int k = (idx < 8) ? (kg * 8 + idx) : (16 + kg * 8 + (idx - 8));
      a.h[idx] = f2bf(attb[(size_t)(m0 + k) * N_ + (r0 + row)]);
    }
#pragma unroll
    for (int j = 0; j < 8; ++j) {
      const unsigned short* Vb = Vt + ((size_t)b * D_ + o0 + j * 16) * (size_t)N_;
      bf16x16 bf = load_frag_rm(Vb, N_, m0, lane);
      acc[j] = wmma_bf16(a.v, bf, acc[j]);
    }
  }

  int col = lane & 15, rhi = (lane >> 4) * 8;
  for (int j = 0; j < 8; ++j)
    for (int i = 0; i < 8; ++i) {
      int n = r0 + rhi + i;
      Y[((size_t)b * N_ + n) * D_ + (o0 + j * 16 + col)] = f2bf(acc[j][i]);
    }
}

// ---------- launch ----------

extern "C" void kernel_launch(void* const* d_in, const int* in_sizes, int n_in,
                              void* d_out, int out_size, void* d_ws, size_t ws_size,
                              hipStream_t stream) {
  const float* x  = (const float*)d_in[0];
  const float* Wk = (const float*)d_in[1];
  const float* Wq = (const float*)d_in[2];
  const float* Wv = (const float*)d_in[3];
  const float* Wo = (const float*)d_in[4];

  float* ybuf = (float*)d_out;                               // [B,N,D] f32
  float* att  = (float*)d_out + (size_t)B_ * N_ * D_;        // [B,N,N] f32

  const size_t XN = (size_t)B_ * N_ * D_;   // 16,777,216 elems
  const size_t WN = (size_t)D_ * D_;        //  1,048,576 elems

  unsigned short* ws  = (unsigned short*)d_ws;
  unsigned short* xb  = ws;  ws += XN;
  unsigned short* Wkb = ws;  ws += WN;
  unsigned short* Wqb = ws;  ws += WN;
  unsigned short* Wvb = ws;  ws += WN;
  unsigned short* Wob = ws;  ws += WN;
  unsigned short* Qb  = ws;  ws += XN;
  unsigned short* Kb  = ws;  ws += XN;
  unsigned short* Vt  = ws;  ws += XN;      // [B, D, N] transposed V
  unsigned short* Yb  = ws;  ws += XN;      // [B, N, D] bf16 y

  // 1) f32 -> bf16
  f32_to_bf16_kernel<<<dim3(4096), 256, 0, stream>>>(x,  xb,  XN);
  f32_to_bf16_kernel<<<dim3(512),  256, 0, stream>>>(Wk, Wkb, WN);
  f32_to_bf16_kernel<<<dim3(512),  256, 0, stream>>>(Wq, Wqb, WN);
  f32_to_bf16_kernel<<<dim3(512),  256, 0, stream>>>(Wv, Wvb, WN);
  f32_to_bf16_kernel<<<dim3(512),  256, 0, stream>>>(Wo, Wob, WN);

  // 2) projections: Mtot=16384, Nc=K=1024 ; grid (16384/64, 1024/128)
  dim3 gg(256, 8);
  gemm_wmma_kernel<<<gg, 256, 0, stream>>>(xb, Wqb, Qb, B_ * N_, D_, D_, QS,   0);
  gemm_wmma_kernel<<<gg, 256, 0, stream>>>(xb, Wkb, Kb, B_ * N_, D_, D_, 1.0f, 0);
  gemm_wmma_kernel<<<gg, 256, 0, stream>>>(xb, Wvb, Vt, B_ * N_, D_, D_, 1.0f, 1);

  // 3) att: zero-fill full N^2, then banded softmax overwrite
  zero_f32_kernel<<<dim3(8192), 256, 0, stream>>>(att, (size_t)B_ * N_ * (size_t)N_);
  band_attn_kernel<<<dim3(N_ / 16, B_), 32, 0, stream>>>(Qb, Kb, att);

  // 4) y = att^T @ V (band-limited)
  av_kernel<<<dim3(N_ / 16, B_), 256, 0, stream>>>(att, Vt, Yb);

  // 5) out = y @ Wo^T (f32)
  gemm_wmma_kernel<<<gg, 256, 0, stream>>>(Yb, Wob, ybuf, B_ * N_, D_, D_, 1.0f, 2);
}
